// DenseAttentionOneHead_26345329393796
// MI455X (gfx1250) — matrix operand order repaired and checked by reading, not compiled, not run on hardware
//
#include <hip/hip_runtime.h>
#include <hip/hip_bf16.h>

// B=2, L=4096, H=1024. Algebraic rewrite: att = h (W^T (h^T h)) -> 3 small GEMMs
// instead of two [L,L]-sized GEMMs (155 GFLOP -> ~39 GFLOP, no 64MB intermediate).
// GEMMs: v_wmma_f32_16x16x32_bf16 fed from LDS tiles staged with
// global_load_async_to_lds_b128 (ASYNCcnt double-buffering).

#define BATCH 2
#define SEQL  4096
#define HID   1024

typedef __attribute__((ext_vector_type(16))) __bf16 v16bf;
typedef __attribute__((ext_vector_type(8)))  __bf16 v8bf;
typedef __attribute__((ext_vector_type(8)))  float  v8f;

// LDS tile geometry: K-chunk = 32 bf16 = 64B data per row, padded to 80B
// (pads fragment reads onto distinct 4-bank groups of the 64 LDS banks).
#define ROW_BYTES    80
#define A_TILE_BYTES (128 * ROW_BYTES)   // 10240 B
#define B_TILE_BYTES (64 * ROW_BYTES)    //  5120 B
#define STAGE_BYTES  (A_TILE_BYTES + B_TILE_BYTES)

// ---------------------------------------------------------------------------
// RoPE: h = x*cos + rotate_half(x)*sin; write h (bf16,[B,L,H]) and hT (bf16,[B,H,L])
// ---------------------------------------------------------------------------
__global__ __launch_bounds__(256) void rope_kernel(
    const float* __restrict__ x, const float* __restrict__ cosT,
    const float* __restrict__ sinT, __bf16* __restrict__ h,
    __bf16* __restrict__ hT)
{
    size_t idx = (size_t)blockIdx.x * blockDim.x + threadIdx.x; // over B*L*H
    int    i   = (int)(idx % HID);
    size_t bl  = idx / HID;
    int    l   = (int)(bl % SEQL);
    int    b   = (int)(bl / SEQL);

    float xv  = x[idx];
    float rot = (i < HID / 2) ? -x[idx + HID / 2] : x[idx - HID / 2];
    float hv  = xv * cosT[(size_t)l * HID + i] + rot * sinT[(size_t)l * HID + i];

    __bf16 hb = (__bf16)hv;
    h[idx] = hb;
    hT[(size_t)b * HID * SEQL + (size_t)i * SEQL + l] = hb;
}

// ---------------------------------------------------------------------------
// Wt[i][j] = W[j][i]  (bf16)
// ---------------------------------------------------------------------------
__global__ __launch_bounds__(256) void wtrans_kernel(
    const float* __restrict__ W, __bf16* __restrict__ Wt)
{
    size_t idx = (size_t)blockIdx.x * blockDim.x + threadIdx.x; // over H*H
    int j = (int)(idx % HID);
    int i = (int)(idx / HID);
    Wt[(size_t)i * HID + j] = (__bf16)W[(size_t)j * HID + i];
}

// ---------------------------------------------------------------------------
// Async staging: block tile A[128 x 32] + Bt[64 x 32] bf16 -> LDS.
// 256 threads x 16B = 4KB per async op; 3 ops per K-chunk. Per-lane LDS byte
// address = low 32 bits of the generic pointer (LDS aperture is in addr[63:32]).
// ---------------------------------------------------------------------------
__device__ __forceinline__ void stage_async(
    const __bf16* __restrict__ A, const __bf16* __restrict__ Bt,
    int lda, int ldb, int mBlk, int nBlk, int k0,
    unsigned char* smemBuf, int tid)
{
    const int row  = tid >> 2;     // 0..63
    const int part = tid & 3;      // 16B chunk within the 64B row
    {   // A rows 0..63
        const __bf16* g = A + (size_t)(mBlk + row) * lda + k0 + part * 8;
        unsigned lds = (unsigned)(size_t)(smemBuf + row * ROW_BYTES + part * 16);
        asm volatile("global_load_async_to_lds_b128 %0, %1, off"
                     :: "v"(lds), "v"(g) : "memory");
    }
    {   // A rows 64..127
        const __bf16* g = A + (size_t)(mBlk + 64 + row) * lda + k0 + part * 8;
        unsigned lds = (unsigned)(size_t)(smemBuf + (64 + row) * ROW_BYTES + part * 16);
        asm volatile("global_load_async_to_lds_b128 %0, %1, off"
                     :: "v"(lds), "v"(g) : "memory");
    }
    {   // Bt rows 0..63
        const __bf16* g = Bt + (size_t)(nBlk + row) * ldb + k0 + part * 8;
        unsigned lds = (unsigned)(size_t)(smemBuf + A_TILE_BYTES + row * ROW_BYTES + part * 16);
        asm volatile("global_load_async_to_lds_b128 %0, %1, off"
                     :: "v"(lds), "v"(g) : "memory");
    }
}

// ---------------------------------------------------------------------------
// WMMA fragment loads from LDS per CDNA5 lane layouts.
// A (16x32): lane row = lane&15, elems {kh*8..+7, kh*8+16..+23}, kh = lane>>4.
// B (32x16 as Bt rows): lane col = lane&15, elems {kh*8..+7, kh*8+16..+23}, kh = lane>>4.
// ---------------------------------------------------------------------------
__device__ __forceinline__ v16bf lds_frag_a(const unsigned char* base, int row0, int lane)
{
    const unsigned char* p = base + (size_t)(row0 + (lane & 15)) * ROW_BYTES
                                  + (lane >> 4) * 16;
    v8bf lo = *(const v8bf*)(p);
    v8bf hi = *(const v8bf*)(p + 32);
    v16bf r;
#pragma unroll
    for (int i = 0; i < 8; ++i) { r[i] = lo[i]; r[i + 8] = hi[i]; }
    return r;
}

__device__ __forceinline__ v16bf lds_frag_b(const unsigned char* base, int row0, int lane)
{
    const unsigned char* p = base + (size_t)(row0 + (lane & 15)) * ROW_BYTES
                                  + (lane >> 4) * 16;
    v8bf lo = *(const v8bf*)(p);
    v8bf hi = *(const v8bf*)(p + 32);
    v16bf r;
#pragma unroll
    for (int i = 0; i < 8; ++i) { r[i] = lo[i]; r[i + 8] = hi[i]; }
    return r;
}

// ---------------------------------------------------------------------------
// GEMM: C[m][n] = sum_k A[m][k] * Bt[n][k]
// 256 threads = 8 waves (4 in M x 2 in N), wave tile 32x32 (2x2 wmma tiles),
// block tile 128(M) x 64(N), K staged in 32-wide double-buffered LDS chunks.
// grid = (N/64, M/128, BATCH).
// ---------------------------------------------------------------------------
template <bool OUT_BF16>
__global__ __launch_bounds__(256) void gemm_bt_kernel(
    const __bf16* __restrict__ A, const __bf16* __restrict__ Bt,
    void* __restrict__ C, int K, int lda, int ldb, int ldc,
    size_t aBatch, size_t bBatch, size_t cBatch)
{
    __shared__ unsigned char smem[2 * STAGE_BYTES];

    const int tid  = threadIdx.x;
    const int lane = tid & 31;
    const int wave = tid >> 5;        // 0..7
    const int wm   = wave >> 1;       // 0..3 (M dir)
    const int wn   = wave & 1;        // 0..1 (N dir)
    const int b    = blockIdx.z;

    A  += (size_t)b * aBatch;
    Bt += (size_t)b * bBatch;

    const int mBlk = blockIdx.y * 128;
    const int nBlk = blockIdx.x * 64;
    const int nk   = K / 32;

    v8f acc00 = {}, acc01 = {}, acc10 = {}, acc11 = {};

    stage_async(A, Bt, lda, ldb, mBlk, nBlk, 0, smem, tid);
    asm volatile("s_wait_asynccnt 0x0" ::: "memory");
    __syncthreads();

    for (int kc = 0; kc < nk; ++kc) {
        unsigned char* cur = smem + (size_t)(kc & 1) * STAGE_BYTES;
        if (kc + 1 < nk) {
            stage_async(A, Bt, lda, ldb, mBlk, nBlk, (kc + 1) * 32,
                        smem + (size_t)((kc + 1) & 1) * STAGE_BYTES, tid);
        }

        const unsigned char* aB = cur;
        const unsigned char* bB = cur + A_TILE_BYTES;
        v16bf a0 = lds_frag_a(aB, wm * 32,      lane);
        v16bf a1 = lds_frag_a(aB, wm * 32 + 16, lane);
        v16bf b0 = lds_frag_b(bB, wn * 32,      lane);
        v16bf b1 = lds_frag_b(bB, wn * 32 + 16, lane);

        acc00 = __builtin_amdgcn_wmma_f32_16x16x32_bf16(false, a0, false, b0,
                                                        (short)0, acc00, false, false);
        acc01 = __builtin_amdgcn_wmma_f32_16x16x32_bf16(false, a0, false, b1,
                                                        (short)0, acc01, false, false);
        acc10 = __builtin_amdgcn_wmma_f32_16x16x32_bf16(false, a1, false, b0,
                                                        (short)0, acc10, false, false);
        acc11 = __builtin_amdgcn_wmma_f32_16x16x32_bf16(false, a1, false, b1,
                                                        (short)0, acc11, false, false);

        asm volatile("s_wait_asynccnt 0x0" ::: "memory");
        __syncthreads();
    }

    // Epilogue: C/D layout -> VGPR r: row = base + r + (lane>=16 ? 8 : 0), col = lane&15
    const int m0   = mBlk + wm * 32;
    const int n0   = nBlk + wn * 32;
    const int rOff = (lane >> 4) * 8;
    const int col0 = lane & 15;
#pragma unroll
    for (int r = 0; r < 8; ++r) {
        int r0 = m0 + r + rOff;
        int r1 = m0 + 16 + r + rOff;
        int c0 = n0 + col0;
        int c1 = n0 + 16 + col0;
        if (OUT_BF16) {
            __bf16* Co = (__bf16*)C + (size_t)b * cBatch;
            Co[(size_t)r0 * ldc + c0] = (__bf16)acc00[r];
            Co[(size_t)r0 * ldc + c1] = (__bf16)acc01[r];
            Co[(size_t)r1 * ldc + c0] = (__bf16)acc10[r];
            Co[(size_t)r1 * ldc + c1] = (__bf16)acc11[r];
        } else {
            float* Co = (float*)C + (size_t)b * cBatch;
            Co[(size_t)r0 * ldc + c0] = acc00[r];
            Co[(size_t)r0 * ldc + c1] = acc01[r];
            Co[(size_t)r1 * ldc + c0] = acc10[r];
            Co[(size_t)r1 * ldc + c1] = acc11[r];
        }
    }
}

// ---------------------------------------------------------------------------
extern "C" void kernel_launch(void* const* d_in, const int* in_sizes, int n_in,
                              void* d_out, int out_size, void* d_ws, size_t ws_size,
                              hipStream_t stream)
{
    const float* x    = (const float*)d_in[0]; // [B,L,H]
    const float* Wq   = (const float*)d_in[1]; // [H,H]
    const float* cosT = (const float*)d_in[2]; // [L,H]
    const float* sinT = (const float*)d_in[3]; // [L,H]
    float*       out  = (float*)d_out;         // [B,L,H]

    // Workspace partition (bf16 buffers), 256B aligned. Total ~42 MB.
    char* ws = (char*)d_ws;
    size_t off = 0;
    auto take = [&](size_t bytes) {
        char* p = ws + off;
        off = (off + bytes + 255) & ~(size_t)255;
        return p;
    };
    __bf16* h  = (__bf16*)take((size_t)BATCH * SEQL * HID * 2); // 16 MB
    __bf16* hT = (__bf16*)take((size_t)BATCH * HID * SEQL * 2); // 16 MB
    __bf16* Wt = (__bf16*)take((size_t)HID * HID * 2);          //  2 MB
    __bf16* G  = (__bf16*)take((size_t)BATCH * HID * HID * 2);  //  4 MB
    __bf16* P  = (__bf16*)take((size_t)BATCH * HID * HID * 2);  //  4 MB
    (void)ws_size;

    // 1) RoPE -> h, hT
    {
        size_t total = (size_t)BATCH * SEQL * HID;
        rope_kernel<<<dim3((unsigned)(total / 256)), dim3(256), 0, stream>>>(
            x, cosT, sinT, h, hT);
    }
    // 2) W^T (bf16)
    {
        size_t total = (size_t)HID * HID;
        wtrans_kernel<<<dim3((unsigned)(total / 256)), dim3(256), 0, stream>>>(Wq, Wt);
    }
    // 3) G = h^T * h : A = hT [HxL], Bt = hT [HxL]; M=N=H, K=L; bf16 out
    gemm_bt_kernel<true><<<dim3(HID / 64, HID / 128, BATCH), dim3(256), 0, stream>>>(
        hT, hT, (void*)G, SEQL, SEQL, SEQL, HID,
        (size_t)HID * SEQL, (size_t)HID * SEQL, (size_t)HID * HID);
    // 4) P = G * W^T-as-Bt : A = G [HxH], Bt = Wt [HxH]; M=N=K=H; bf16 out
    gemm_bt_kernel<true><<<dim3(HID / 64, HID / 128, BATCH), dim3(256), 0, stream>>>(
        G, Wt, (void*)P, HID, HID, HID, HID,
        (size_t)HID * HID, (size_t)0, (size_t)HID * HID);
    // 5) out = h * P-as-Bt : A = h [LxH], Bt = P [HxH]; M=L, N=H, K=H; f32 out
    gemm_bt_kernel<false><<<dim3(HID / 64, SEQL / 128, BATCH), dim3(256), 0, stream>>>(
        h, P, (void*)out, HID, HID, HID, HID,
        (size_t)SEQL * HID, (size_t)HID * HID, (size_t)SEQL * HID);
}
